// HeadWisePhybridAttention_8907762172232
// MI455X (gfx1250) — matrix operand order repaired and checked by reading, not compiled, not run on hardware
//
#include <hip/hip_runtime.h>
#include <stdint.h>

// ---------------------------------------------------------------------------
// Problem constants (match reference)
// ---------------------------------------------------------------------------
#define BATCH 2
#define LSEQ  2048
#define HDIM  2048
#define NKV   8
#define HD    128
#define H2    8           // NH/2
#define KVDIM 1024        // NKV*HD
#define ROWS  4096        // BATCH*LSEQ

typedef __attribute__((ext_vector_type(16))) _Float16 v16h;
typedef __attribute__((ext_vector_type(8)))  _Float16 v8h;
typedef __attribute__((ext_vector_type(8)))  float    v8f;

#define WMMA_F16(A, B, C) \
  __builtin_amdgcn_wmma_f32_16x16x32_f16(false, (A), false, (B), (short)0, (C), false, false)

// ---------------------------------------------------------------------------
// CDNA5 async global->LDS copy (16 bytes per lane, tracked by ASYNCcnt).
// ---------------------------------------------------------------------------
static __device__ __forceinline__ unsigned int lds_off(const void* p) {
  return (unsigned int)(uintptr_t)p;
}
static __device__ __forceinline__ void async_cp16(unsigned int lds, const void* g) {
  asm volatile("global_load_async_to_lds_b128 %0, %1, off"
               :: "v"(lds), "v"((unsigned long long)(uintptr_t)g)
               : "memory");
}
static __device__ __forceinline__ void wait_async0() {
  asm volatile("s_wait_asynccnt 0x0" ::: "memory");
}

// ---------------------------------------------------------------------------
// WMMA fragment loaders (CDNA5 16x16x32 f16 layouts, wave32)
// A (16xK slice, row-major, lda halves): lane l holds row (l&15);
//   elems 0..7 -> k=(l>>4)*8+e ; elems 8..15 -> k=16+(l>>4)*8+(e-8)
// B fed K-contiguous per column (ldn halves): lane l holds col (l&15);
//   elems 0..7 -> k=(l>>4)*8+e ; elems 8..15 -> k=16+(l>>4)*8+(e-8)
// ---------------------------------------------------------------------------
static __device__ __forceinline__ v16h frag_a(const _Float16* base, int lda) {
  const int lane = threadIdx.x & 31;
  const _Float16* p = base + (lane & 15) * lda + ((lane >> 4) << 3);
  v8h lo = *(const v8h*)p;
  v8h hi = *(const v8h*)(p + 16);
  v16h r;
#pragma unroll
  for (int i = 0; i < 8; ++i) { r[i] = lo[i]; r[i + 8] = hi[i]; }
  return r;
}

static __device__ __forceinline__ v16h frag_b(const _Float16* base, int ldn) {
  const int lane = threadIdx.x & 31;
  const _Float16* p = base + (lane & 15) * ldn + ((lane >> 4) << 3);
  v8h lo = *(const v8h*)p;
  v8h hi = *(const v8h*)(p + 16);
  v16h r;
#pragma unroll
  for (int i = 0; i < 8; ++i) { r[i] = lo[i]; r[i + 8] = hi[i]; }
  return r;
}

// ---------------------------------------------------------------------------
// small helpers
// ---------------------------------------------------------------------------
__global__ __launch_bounds__(256) void cvt_f32_f16(const float* __restrict__ x,
                                                   _Float16* __restrict__ y, int n) {
  int i = blockIdx.x * 256 + threadIdx.x;
  if (i < n) y[i] = (_Float16)x[i];
}

__global__ __launch_bounds__(256) void fill_zero(float* __restrict__ p, int n) {
  int i = blockIdx.x * 256 + threadIdx.x;
  if (i < n) p[i] = 0.f;
}

// f32 [R][C] -> f16 transposed [C][R] (LDS-tiled, one-time weight prep)
__global__ __launch_bounds__(256)
void transpose_cvt(const float* __restrict__ src, _Float16* __restrict__ dst,
                   int R, int C) {
  __shared__ float tile[32][33];
  int c0 = blockIdx.x * 32, r0 = blockIdx.y * 32;
  int x = threadIdx.x & 31, y = threadIdx.x >> 5;
#pragma unroll
  for (int j = 0; j < 32; j += 8)
    tile[y + j][x] = src[(size_t)(r0 + y + j) * C + c0 + x];
  __syncthreads();
#pragma unroll
  for (int j = 0; j < 32; j += 8)
    dst[(size_t)(c0 + y + j) * R + r0 + x] = (_Float16)tile[x][y + j];
}

// ---------------------------------------------------------------------------
// GEMM: C[M,N] = A[M,K] @ B[K,N] + bias, B pre-transposed BT[N][K].
// 128x128 block tile, 256 threads (8 waves), wave = 32x64 sub-tile.
// Double-buffered LDS, async global->LDS staging, ONE barrier per 64-K step,
// 16 WMMAs per wave per step.
// ---------------------------------------------------------------------------
__global__ __launch_bounds__(256)
void wmma_gemm(const _Float16* __restrict__ A, const _Float16* __restrict__ BT,
               const float* __restrict__ bias, float* __restrict__ C,
               int M, int N, int Kd) {
  __shared__ __align__(32) _Float16 As[2][128 * 72];   // [m][k0..63], lda=72
  __shared__ __align__(32) _Float16 Bs[2][128 * 72];   // [n][k0..63], ldk=72
  const int n0 = blockIdx.x * 128;
  const int m0 = blockIdx.y * 128;
  const int t = threadIdx.x;
  const int wave = t >> 5, lane = t & 31;
  const int wm = (wave >> 1) * 32, wn = (wave & 1) * 64;
  const int r = t >> 1, seg = (t & 1) * 32;

  const _Float16* gA = A + (size_t)(m0 + r) * Kd + seg;
  const _Float16* gB = BT + (size_t)(n0 + r) * Kd + seg;

  auto stage = [&](int buf, int k0) {
    unsigned la = lds_off(&As[buf][r * 72 + seg]);
    unsigned lb = lds_off(&Bs[buf][r * 72 + seg]);
    const _Float16* pa = gA + k0;
    const _Float16* pb = gB + k0;
#pragma unroll
    for (int u = 0; u < 4; ++u) {
      async_cp16(la + u * 16, pa + u * 8);
      async_cp16(lb + u * 16, pb + u * 8);
    }
    __builtin_prefetch(pa + 64, 0, 3);
    __builtin_prefetch(pb + 64, 0, 3);
  };

  v8f acc[2][4] = {};
  const int nk = Kd >> 6;
  stage(0, 0);
  for (int kt = 0; kt < nk; ++kt) {
    const int cur = kt & 1;
    wait_async0();            // own tile-kt copies complete
    __syncthreads();          // everyone's copies complete; prior reads done
    if (kt + 1 < nk) stage(cur ^ 1, (kt + 1) * 64);
#pragma unroll
    for (int ks = 0; ks < 2; ++ks) {
      v16h a0 = frag_a(&As[cur][(wm + 0) * 72 + ks * 32], 72);
      v16h a1 = frag_a(&As[cur][(wm + 16) * 72 + ks * 32], 72);
#pragma unroll
      for (int nt = 0; nt < 4; ++nt) {
        v16h b = frag_b(&Bs[cur][(wn + nt * 16) * 72 + ks * 32], 72);
        acc[0][nt] = WMMA_F16(a0, b, acc[0][nt]);
        acc[1][nt] = WMMA_F16(a1, b, acc[1][nt]);
      }
    }
  }

  const int ci = lane & 15, hi = lane >> 4;
#pragma unroll
  for (int mt = 0; mt < 2; ++mt) {
#pragma unroll
    for (int nt = 0; nt < 4; ++nt) {
      int col = n0 + wn + nt * 16 + ci;
      float bv = bias[col];
#pragma unroll
      for (int v = 0; v < 8; ++v) {
        int row = m0 + wm + mt * 16 + hi * 8 + v;
        C[(size_t)row * N + col] = acc[mt][nt][v] + bv;
      }
    }
  }
}

// ---------------------------------------------------------------------------
// RoPE on softmax-half q and k, f16 per-head [B,h,L,HD]; v converted to f16
// in transposed [B,kv,HD,L] layout (attention V tiles become straight copies).
// ---------------------------------------------------------------------------
__global__ __launch_bounds__(256)
void rope_prep(const float* __restrict__ Q, const float* __restrict__ Kf,
               const float* __restrict__ Vf, _Float16* __restrict__ QS,
               _Float16* __restrict__ KS, _Float16* __restrict__ VST) {
  int idx = blockIdx.x * 256 + threadIdx.x;
  if (idx >= BATCH * 8 * LSEQ * HD) return;
  int d = idx & 127;
  int l = (idx >> 7) & (LSEQ - 1);
  int h = (idx >> 18) & 7;
  int b = idx >> 21;

  int fi = d & 63;
  float inv = __expf(-(2.f * (float)fi / 128.f) * 13.815510558f); // 1e6^(-2fi/128)
  float ang = (float)l * inv;
  float c = __cosf(ang), s = __sinf(ang);

  size_t dst = (((size_t)(b * 8 + h)) * LSEQ + l) * HD + d;

  const float* qrow = Q + ((size_t)b * LSEQ + l) * HDIM + h * HD;
  float qv = qrow[d];
  float qo = (d < 64) ? -qrow[d + 64] : qrow[d - 64];
  QS[dst] = (_Float16)(qv * c + qo * s);

  const float* krow = Kf + ((size_t)b * LSEQ + l) * KVDIM + h * HD;
  float kv = krow[d];
  float ko = (d < 64) ? -krow[d + 64] : krow[d - 64];
  KS[dst] = (_Float16)(kv * c + ko * s);

  float vv = Vf[((size_t)b * LSEQ + l) * KVDIM + h * HD + d];
  VST[(((size_t)(b * 8 + h)) * HD + d) * LSEQ + l] = (_Float16)vv;
}

// ---------------------------------------------------------------------------
// Causal depthwise conv (K=4) + SiLU for the linear half.
// ---------------------------------------------------------------------------
__global__ __launch_bounds__(256)
void conv_prep(const float* __restrict__ Q, const float* __restrict__ Kf,
               const float* __restrict__ qcw, const float* __restrict__ qcb,
               const float* __restrict__ kcw, const float* __restrict__ kcb,
               _Float16* __restrict__ QT, _Float16* __restrict__ KC) {
  int idx = blockIdx.x * 256 + threadIdx.x;
  if (idx >= BATCH * 1024 * LSEQ) return;
  int l = idx & (LSEQ - 1);
  int c = (idx >> 11) & 1023;
  int b = idx >> 21;

  float aq = qcb[c], ak = kcb[c];
#pragma unroll
  for (int i = 0; i < 4; ++i) {
    int ll = l - 3 + i;
    if (ll >= 0) {
      aq += Q[((size_t)b * LSEQ + ll) * HDIM + 1024 + c] * qcw[c * 4 + i];
      ak += Kf[((size_t)b * LSEQ + ll) * KVDIM + c] * kcw[c * 4 + i];
    }
  }
  float sq = aq / (1.f + __expf(-aq));
  float sk = ak / (1.f + __expf(-ak));
  int h = c >> 7, d = c & 127;
  size_t dst = (((size_t)(b * 8 + h)) * LSEQ + l) * HD + d;
  QT[dst] = (_Float16)sq;
  KC[dst] = (_Float16)sk;
}

// ---------------------------------------------------------------------------
// Flash softmax attention: block = (64-row q tile, head, batch), 128 thr.
// Double-buffered K / V^T tiles via async global->LDS, one barrier per tile.
// Pw is per-wave scratch (same-wave LDS ordering, no barrier needed).
// ---------------------------------------------------------------------------
__global__ __launch_bounds__(128)
void attn_softmax(const _Float16* __restrict__ QS, const _Float16* __restrict__ KS,
                  const _Float16* __restrict__ VST, _Float16* __restrict__ CAT) {
  __shared__ __align__(32) _Float16 Kt[2][64 * 128];   // [key][d]
  __shared__ __align__(32) _Float16 VtT[2][128 * 64];  // [d][key]
  __shared__ __align__(32) _Float16 Pw[4][16 * 64];
  const int q0 = blockIdx.x * 64;
  const int h = blockIdx.y;
  const int b = blockIdx.z;
  const int t = threadIdx.x, wave = t >> 5, lane = t & 31;
  const int kv = h >> 1;
  const int ci = lane & 15, hi16 = lane >> 4;

  const _Float16* Qbase = QS + (((size_t)(b * 8 + h)) * LSEQ + q0 + wave * 16) * HD;
  v16h qf[4];
#pragma unroll
  for (int f = 0; f < 4; ++f) qf[f] = frag_a(Qbase + f * 32, HD);

  float mi[8], li[8];
  v8f O[8] = {};
#pragma unroll
  for (int v = 0; v < 8; ++v) { mi[v] = -1e30f; li[v] = 0.f; }

  const _Float16* Kbase = KS + ((size_t)(b * 8 + kv)) * LSEQ * HD;
  const _Float16* VTbase = VST + ((size_t)(b * 8 + kv)) * HD * LSEQ;
  const int kr = t >> 1, kc0 = (t & 1) * 64;

  auto stage = [&](int buf, int j0) {
    unsigned laK = lds_off(&Kt[buf][kr * 128 + kc0]);
    unsigned laV = lds_off(&VtT[buf][t * 64]);
    const _Float16* gK = Kbase + (size_t)(j0 + kr) * HD + kc0;
    const _Float16* gV = VTbase + (size_t)t * LSEQ + j0;
#pragma unroll
    for (int u = 0; u < 8; ++u) {
      async_cp16(laK + u * 16, gK + u * 8);
      async_cp16(laV + u * 16, gV + u * 8);
    }
  };

  const int nj = (q0 >> 6) + 1;
  stage(0, 0);
  for (int jt = 0; jt < nj; ++jt) {
    const int cur = jt & 1;
    const int j0 = jt * 64;
    wait_async0();
    __syncthreads();
    if (jt + 1 < nj) stage(cur ^ 1, (jt + 1) * 64);

    // S = Q @ K^T
    v8f S[4] = {};
#pragma unroll
    for (int nt = 0; nt < 4; ++nt)
#pragma unroll
      for (int f = 0; f < 4; ++f)
        S[nt] = WMMA_F16(qf[f], frag_b(&Kt[cur][(nt * 16) * 128 + f * 32], 128), S[nt]);

    const float scale = 0.08838834764831845f; // 1/sqrt(128)
    const bool diag = (jt == nj - 1);
#pragma unroll
    for (int nt = 0; nt < 4; ++nt)
#pragma unroll
      for (int v = 0; v < 8; ++v) {
        float s = S[nt][v] * scale;
        if (diag) {
          int rg = q0 + wave * 16 + hi16 * 8 + v;
          int kg = j0 + nt * 16 + ci;
          if (kg > rg) s = -1e30f;
        }
        S[nt][v] = s;
      }

    // flash update per row
#pragma unroll
    for (int v = 0; v < 8; ++v) {
      float mx = fmaxf(fmaxf(S[0][v], S[1][v]), fmaxf(S[2][v], S[3][v]));
#pragma unroll
      for (int off = 1; off < 16; off <<= 1) mx = fmaxf(mx, __shfl_xor(mx, off, 32));
      float mnew = fmaxf(mi[v], mx);
      float sc = __expf(mi[v] - mnew);
      mi[v] = mnew;
      float rs = 0.f;
#pragma unroll
      for (int nt = 0; nt < 4; ++nt) { float p = __expf(S[nt][v] - mnew); S[nt][v] = p; rs += p; }
#pragma unroll
      for (int off = 1; off < 16; off <<= 1) rs += __shfl_xor(rs, off, 32);
      li[v] = li[v] * sc + rs;
#pragma unroll
      for (int n2 = 0; n2 < 8; ++n2) O[n2][v] *= sc;
    }

    // P -> per-wave LDS scratch (C-tile layout -> A-fragment layout)
    _Float16* pw = Pw[wave];
#pragma unroll
    for (int nt = 0; nt < 4; ++nt)
#pragma unroll
      for (int v = 0; v < 8; ++v)
        pw[(hi16 * 8 + v) * 64 + nt * 16 + ci] = (_Float16)S[nt][v];

    // O += P @ V  (same-wave LDS RAW: in-order, no barrier)
#pragma unroll
    for (int ks = 0; ks < 2; ++ks) {
      v16h pa = frag_a(pw + ks * 32, 64);
#pragma unroll
      for (int nt = 0; nt < 8; ++nt)
        O[nt] = WMMA_F16(pa, frag_b(&VtT[cur][(nt * 16) * 64 + ks * 32], 64), O[nt]);
    }
  }

#pragma unroll
  for (int v = 0; v < 8; ++v) {
    float invl = 1.f / li[v];
    int rg = q0 + wave * 16 + hi16 * 8 + v;
    _Float16* crow = CAT + ((size_t)b * LSEQ + rg) * HDIM + h * HD;
#pragma unroll
    for (int nt = 0; nt < 8; ++nt) crow[nt * 16 + ci] = (_Float16)(O[nt][v] * invl);
  }
}

// ---------------------------------------------------------------------------
// Linear (lightning) attention with per-head decay exp(-slope*(i-j)).
// ---------------------------------------------------------------------------
__global__ __launch_bounds__(128)
void attn_linear(const _Float16* __restrict__ QT, const _Float16* __restrict__ KC,
                 const _Float16* __restrict__ VST, const float* __restrict__ slope_rate,
                 float* __restrict__ OT) {
  __shared__ __align__(32) _Float16 Kt[2][64 * 128];
  __shared__ __align__(32) _Float16 VtT[2][128 * 64];
  __shared__ __align__(32) _Float16 Pw[4][16 * 64];
  const int q0 = blockIdx.x * 64;
  const int hh = blockIdx.y;        // linear head 0..7 (global head 8+hh)
  const int b = blockIdx.z;
  const int t = threadIdx.x, wave = t >> 5, lane = t & 31;
  const int kv = (8 + hh) >> 1;     // kv head for k_lin / v_lin
  const int ci = lane & 15, hi16 = lane >> 4;
  const float slope = slope_rate[hh];

  const _Float16* Qbase = QT + (((size_t)(b * 8 + hh)) * LSEQ + q0 + wave * 16) * HD;
  v16h qf[4];
#pragma unroll
  for (int f = 0; f < 4; ++f) qf[f] = frag_a(Qbase + f * 32, HD);

  v8f O[8] = {};
  const _Float16* Kbase = KC + ((size_t)(b * 8 + kv)) * LSEQ * HD;
  const _Float16* VTbase = VST + ((size_t)(b * 8 + kv)) * HD * LSEQ;
  const int kr = t >> 1, kc0 = (t & 1) * 64;

  auto stage = [&](int buf, int j0) {
    unsigned laK = lds_off(&Kt[buf][kr * 128 + kc0]);
    unsigned laV = lds_off(&VtT[buf][t * 64]);
    const _Float16* gK = Kbase + (size_t)(j0 + kr) * HD + kc0;
    const _Float16* gV = VTbase + (size_t)t * LSEQ + j0;
#pragma unroll
    for (int u = 0; u < 8; ++u) {
      async_cp16(laK + u * 16, gK + u * 8);
      async_cp16(laV + u * 16, gV + u * 8);
    }
  };

  const int nj = (q0 >> 6) + 1;
  stage(0, 0);
  for (int jt = 0; jt < nj; ++jt) {
    const int cur = jt & 1;
    const int j0 = jt * 64;
    wait_async0();
    __syncthreads();
    if (jt + 1 < nj) stage(cur ^ 1, (jt + 1) * 64);

    v8f S[4] = {};
#pragma unroll
    for (int nt = 0; nt < 4; ++nt)
#pragma unroll
      for (int f = 0; f < 4; ++f)
        S[nt] = WMMA_F16(qf[f], frag_b(&Kt[cur][(nt * 16) * 128 + f * 32], 128), S[nt]);

    // causal decay weight
#pragma unroll
    for (int nt = 0; nt < 4; ++nt)
#pragma unroll
      for (int v = 0; v < 8; ++v) {
        int rg = q0 + wave * 16 + hi16 * 8 + v;
        int kg = j0 + nt * 16 + ci;
        float w = (kg <= rg) ? __expf(-slope * (float)(rg - kg)) : 0.f;
        S[nt][v] *= w;
      }

    _Float16* pw = Pw[wave];
#pragma unroll
    for (int nt = 0; nt < 4; ++nt)
#pragma unroll
      for (int v = 0; v < 8; ++v)
        pw[(hi16 * 8 + v) * 64 + nt * 16 + ci] = (_Float16)S[nt][v];

#pragma unroll
    for (int ks = 0; ks < 2; ++ks) {
      v16h pa = frag_a(pw + ks * 32, 64);
#pragma unroll
      for (int nt = 0; nt < 8; ++nt)
        O[nt] = WMMA_F16(pa, frag_b(&VtT[cur][(nt * 16) * 64 + ks * 32], 64), O[nt]);
    }
  }

#pragma unroll
  for (int v = 0; v < 8; ++v) {
    int rg = q0 + wave * 16 + hi16 * 8 + v;
    float* orow = OT + ((size_t)b * LSEQ + rg) * 1024 + hh * HD;
#pragma unroll
    for (int nt = 0; nt < 8; ++nt) orow[nt * 16 + ci] = O[nt][v];
  }
}

// ---------------------------------------------------------------------------
// SimpleRMSNorm over 1024 features (f32) -> f16 into concat cols 1024..2047
// ---------------------------------------------------------------------------
__global__ __launch_bounds__(256)
void rmsnorm_cat(const float* __restrict__ OT, _Float16* __restrict__ CAT) {
  __shared__ float red[256];
  int row = blockIdx.x;
  const float* x = OT + (size_t)row * 1024;
  float ss = 0.f;
  for (int i = threadIdx.x; i < 1024; i += 256) { float v = x[i]; ss += v * v; }
  red[threadIdx.x] = ss;
  __syncthreads();
  for (int s = 128; s > 0; s >>= 1) {
    if (threadIdx.x < s) red[threadIdx.x] += red[threadIdx.x + s];
    __syncthreads();
  }
  float r = rsqrtf(red[0] / 1024.f + 1e-6f);
  for (int i = threadIdx.x; i < 1024; i += 256)
    CAT[(size_t)row * HDIM + 1024 + i] = (_Float16)(x[i] * r);
}

// ---------------------------------------------------------------------------
// Host-side orchestration
// ---------------------------------------------------------------------------
extern "C" void kernel_launch(void* const* d_in, const int* in_sizes, int n_in,
                              void* d_out, int out_size, void* d_ws, size_t ws_size,
                              hipStream_t stream) {
  (void)in_sizes; (void)n_in; (void)out_size; (void)ws_size;
  const float* hs    = (const float*)d_in[0];
  const float* slope = (const float*)d_in[3];
  const float* Wq    = (const float*)d_in[5];
  const float* bq    = (const float*)d_in[6];
  const float* Wk    = (const float*)d_in[7];
  const float* bk    = (const float*)d_in[8];
  const float* Wv    = (const float*)d_in[9];
  const float* bv    = (const float*)d_in[10];
  const float* Wo    = (const float*)d_in[11];
  const float* qcw   = (const float*)d_in[12];
  const float* qcb   = (const float*)d_in[13];
  const float* kcw   = (const float*)d_in[14];
  const float* kcb   = (const float*)d_in[15];

  char* ws = (char*)d_ws;
  size_t o = 0;
  _Float16* XH  = (_Float16*)(ws + o); o += (size_t)ROWS * HDIM * 2;     // A (row-major)
  _Float16* WQT = (_Float16*)(ws + o); o += (size_t)HDIM * HDIM * 2;     // [N][K]
  _Float16* WKT = (_Float16*)(ws + o); o += (size_t)KVDIM * HDIM * 2;
  _Float16* WVT = (_Float16*)(ws + o); o += (size_t)KVDIM * HDIM * 2;
  _Float16* WOT = (_Float16*)(ws + o); o += (size_t)HDIM * HDIM * 2;
  float*    Qf  = (float*)(ws + o);    o += (size_t)ROWS * HDIM * 4;
  float*    Kf  = (float*)(ws + o);    o += (size_t)ROWS * KVDIM * 4;
  float*    Vf  = (float*)(ws + o);    o += (size_t)ROWS * KVDIM * 4;
  _Float16* QSh = (_Float16*)(ws + o); o += (size_t)BATCH * 8 * LSEQ * HD * 2;
  _Float16* KSh = (_Float16*)(ws + o); o += (size_t)BATCH * 8 * LSEQ * HD * 2;
  _Float16* VSTh= (_Float16*)(ws + o); o += (size_t)BATCH * 8 * HD * LSEQ * 2;
  _Float16* QTh = (_Float16*)(ws + o); o += (size_t)BATCH * 8 * LSEQ * HD * 2;
  _Float16* KCh = (_Float16*)(ws + o); o += (size_t)BATCH * 8 * LSEQ * HD * 2;
  _Float16* CATh= (_Float16*)(ws + o); o += (size_t)ROWS * HDIM * 2;
  float*    OTf = (float*)(ws + o);    o += (size_t)ROWS * 1024 * 4;
  float*    ZB  = (float*)(ws + o);    o += (size_t)HDIM * 4;            // zero bias

  // 1) f16 staging: X row-major; weights transposed to [N][K]; zero bias
  cvt_f32_f16<<<(ROWS * HDIM) / 256, 256, 0, stream>>>(hs, XH, ROWS * HDIM);
  transpose_cvt<<<dim3(HDIM / 32, HDIM / 32), 256, 0, stream>>>(Wq, WQT, HDIM, HDIM);
  transpose_cvt<<<dim3(KVDIM / 32, HDIM / 32), 256, 0, stream>>>(Wk, WKT, HDIM, KVDIM);
  transpose_cvt<<<dim3(KVDIM / 32, HDIM / 32), 256, 0, stream>>>(Wv, WVT, HDIM, KVDIM);
  transpose_cvt<<<dim3(HDIM / 32, HDIM / 32), 256, 0, stream>>>(Wo, WOT, HDIM, HDIM);
  fill_zero<<<HDIM / 256, 256, 0, stream>>>(ZB, HDIM);

  // 2) QKV projections (WMMA, double-buffered async-staged tiles)
  wmma_gemm<<<dim3(HDIM / 128, ROWS / 128), 256, 0, stream>>>(XH, WQT, bq, Qf, ROWS, HDIM, HDIM);
  wmma_gemm<<<dim3(KVDIM / 128, ROWS / 128), 256, 0, stream>>>(XH, WKT, bk, Kf, ROWS, KVDIM, HDIM);
  wmma_gemm<<<dim3(KVDIM / 128, ROWS / 128), 256, 0, stream>>>(XH, WVT, bv, Vf, ROWS, KVDIM, HDIM);

  // 3) element-wise prep (RoPE also writes V transposed)
  rope_prep<<<(BATCH * 8 * LSEQ * HD) / 256, 256, 0, stream>>>(Qf, Kf, Vf, QSh, KSh, VSTh);
  conv_prep<<<(BATCH * 1024 * LSEQ) / 256, 256, 0, stream>>>(Qf, Kf, qcw, qcb, kcw, kcb, QTh, KCh);

  // 4) attention halves (WMMA)
  attn_softmax<<<dim3(LSEQ / 64, H2, BATCH), 128, 0, stream>>>(QSh, KSh, VSTh, CATh);
  attn_linear<<<dim3(LSEQ / 64, H2, BATCH), 128, 0, stream>>>(QTh, KCh, VSTh, slope, OTf);

  // 5) RMSNorm of linear half into concat buffer
  rmsnorm_cat<<<ROWS, 256, 0, stream>>>(OTf, CATh);

  // 6) output projection (WMMA) -> f32 d_out
  wmma_gemm<<<dim3(HDIM / 128, ROWS / 128), 256, 0, stream>>>(CATh, WOT, ZB,
                                                              (float*)d_out, ROWS, HDIM, HDIM);
}
